// NeRFTrainer_35991825940594
// MI455X (gfx1250) — matrix operand order repaired and checked by reading, not compiled, not run on hardware
//
#include <hip/hip_runtime.h>
#include <hip/hip_bf16.h>

// ---------------------------------------------------------------------------
// NeRF forward on gfx1250 (MI455X): f16 WMMA batched-MLP + volume rendering.
// B=1024 rays, NC=64 coarse, NF=128 fine, HID=128, HDIR=64.
//
// Weights are transposed+converted to f16 ONCE (pack_weights_kernel) into a
// [n][k] K-padded layout. MLP blocks process 128 samples with LDS-resident
// activations; weight chunks are double-buffered in LDS so staging overlaps
// WMMA compute; A/B fragments are register-hoisted for maximal LDS reuse.
// All LDS accesses are element offsets into the dynamic-LDS symbol so the
// compiler keeps them in addrspace(3) (no generic-pointer null-check VALU).
// ---------------------------------------------------------------------------

#define DEV __device__ __forceinline__

typedef __attribute__((ext_vector_type(16))) _Float16 v16h;
typedef __attribute__((ext_vector_type(8)))  _Float16 h8;
typedef __attribute__((ext_vector_type(8)))  float    v8f;

// packed-weight segment offsets (halves) within one MLP's pack
#define OFF_W0   0        // 128 x 64   (K 63 padded to 64)
#define OFF_W1   8192     // 128 x 128
#define OFF_W2   24576    // 128 x 128
#define OFF_W3   40960    // 128 x 128
#define OFF_WF   57344    // 128 x 128  (Wfeat)
#define OFF_WD   73728    // 64  x 160  (Wdir, K 155 padded to 160)
#define MLP_PACK 83968    // halves per MLP

// dynamic-LDS layout (element offsets, _Float16 units)
#define SMEM_A   0        // 128 x 128 activations (ping)
#define SMEM_B   16384    // 128 x 128 activations (pong)
#define SMEM_D   32768    // 128 x 32  dirs encoding (27 + pad)
#define SMEM_W   36864    // 2 x 8192  double-buffered W^T chunks (16384)
#define SMEM_HALVES 53248 // -> 106496 bytes

// --------------------------- WMMA fragment helpers ---------------------------
// A-fragment (16x32 f16): lane L<16 holds row M=L, K chunks {0..7, 16..23};
// lane L>=16 holds row M=L-16, K chunks {8..15, 24..31}.   (ISA 7.12.2)
// B-fragment (32x16 f16): lane L<16 holds col N=L, K chunks {0..7, 16..23};
// lane L>=16 holds col N=L-16, K chunks {8..15, 24..31}.  Weights live transposed: wT[n][k].
// C/D (16x16 f32): elem v -> row = (lane>=16 ? 8:0)+v, col = lane&15.

DEV v16h load_afrag(int aOff, int ldA, int row, int kOff) {
  extern __shared__ _Float16 smem_h[];
  const int lane = threadIdx.x & 31;
  const int hi   = lane >> 4;
  const int base = aOff + (row + (lane & 15)) * ldA + kOff + (hi ? 8 : 0);
  h8 a0 = *(const h8*)&smem_h[base];
  h8 a1 = *(const h8*)&smem_h[base + 16];
  return __builtin_shufflevector(a0, a1, 0,1,2,3,4,5,6,7,8,9,10,11,12,13,14,15);
}

DEV v16h load_bfrag(int wOff, int ldW, int nLocal, int kOff) {
  extern __shared__ _Float16 smem_h[];
  const int lane = threadIdx.x & 31;
  const int hi   = lane >> 4;
  const int base = wOff + (nLocal + (lane & 15)) * ldW + kOff + (hi ? 8 : 0);
  h8 b0 = *(const h8*)&smem_h[base];
  h8 b1 = *(const h8*)&smem_h[base + 16];
  return __builtin_shufflevector(b0, b1, 0,1,2,3,4,5,6,7,8,9,10,11,12,13,14,15);
}

template<bool RELU>
DEV void store_tile(v8f c, int outOff, int ldO, int rowBase, int col0) {
  extern __shared__ _Float16 smem_h[];
  const int lane = threadIdx.x & 31;
  const int hi   = lane >> 4;
  const int col  = col0 + (lane & 15);
#pragma unroll
  for (int v = 0; v < 8; ++v) {
    float x = c[v];
    if (RELU) x = x > 0.0f ? x : 0.0f;
    smem_h[outOff + (rowBase + (hi ? 8 : 0) + v) * ldO + col] = (_Float16)x;
  }
}

// contiguous b128 copy of one 64-col weight chunk (64*KP halves) into LDS
template<int KP>
DEV void stage_chunk(int dstOff, const _Float16* __restrict__ src) {
  extern __shared__ _Float16 smem_h[];
  constexpr int NV = (64 * KP) / 8;          // h8 vectors
  const int tid = threadIdx.x;
  const h8* s = (const h8*)src;
#pragma unroll
  for (int i = 0; i < NV / 128; ++i)
    *(h8*)&smem_h[dstOff + (tid + i * 128) * 8] = s[tid + i * 128];
}

// ------------------------------ dense layer ----------------------------------
// 128-sample LDS tile, 128 threads (4 waves x 32 rows). Input may be split
// across two LDS regions (A1: K1T*32 cols, A2: K2T*32 cols) to support
// concat(feat, dirs_enc) without a copy. Weight chunks (64 cols) are double-
// buffered: chunk ph+1 is staged while chunk ph is consumed by WMMAs.
// Bias is read straight from global (cache-hot, one scalar per tile).
template<int K1T, int K2T, int N, bool RELU>
DEV void dense_layer(int a1Off, int ldA1, int a2Off, int ldA2,
                     const _Float16* __restrict__ wTg,
                     const float* __restrict__ bglob,
                     int outOff, int ldO) {
  constexpr int KT     = K1T + K2T;
  constexpr int KP     = KT * 32;
  constexpr int CHUNK  = 64 * KP;            // halves per staged chunk
  constexpr int PHASES = N / 64;
  const int tid  = threadIdx.x;
  const int wave = tid >> 5;
  const int rowBase = wave * 32;

  stage_chunk<KP>(SMEM_W, wTg);              // chunk 0
  __syncthreads();

#pragma unroll
  for (int ph = 0; ph < PHASES; ++ph) {
    const int curOff = SMEM_W + (ph & 1) * CHUNK;
    if (ph + 1 < PHASES)                     // overlap next-chunk staging
      stage_chunk<KP>(SMEM_W + ((ph + 1) & 1) * CHUNK, wTg + (ph + 1) * CHUNK);

    // hoist A-fragments: one per (row-tile, k-tile), reused by 4 col-tiles
    v16h aF[2][KT];
#pragma unroll
    for (int mt = 0; mt < 2; ++mt) {
#pragma unroll
      for (int kt = 0; kt < K1T; ++kt)
        aF[mt][kt] = load_afrag(a1Off, ldA1, rowBase + mt * 16, kt * 32);
      if constexpr (K2T > 0) {
#pragma unroll
        for (int kt = 0; kt < K2T; ++kt)
          aF[mt][K1T + kt] = load_afrag(a2Off, ldA2, rowBase + mt * 16, kt * 32);
      }
    }
#pragma unroll
    for (int nt = 0; nt < 64; nt += 16) {
      // hoist B-fragments: one per k-tile, reused by both row-tiles
      v16h bF[KT];
#pragma unroll
      for (int kt = 0; kt < KT; ++kt)
        bF[kt] = load_bfrag(curOff, KP, nt, kt * 32);
      const float bv = bglob[ph * 64 + nt + (tid & 15)];
#pragma unroll
      for (int mt = 0; mt < 2; ++mt) {
        v8f c = {bv, bv, bv, bv, bv, bv, bv, bv};
#pragma unroll
        for (int kt = 0; kt < KT; ++kt)
          c = __builtin_amdgcn_wmma_f32_16x16x32_f16(false, aF[mt][kt], false,
                                                     bF[kt], (short)0, c,
                                                     false, false);
        store_tile<RELU>(c, outOff, ldO, rowBase + mt * 16, ph * 64 + nt);
      }
    }
    __syncthreads();
  }
}

// ----------------------- one-time weight pack kernel ------------------------
// Transpose + f32->f16 both MLPs into [n][k] K-padded layout. 12 blocks:
// blockIdx = mlp*6 + segment. Reads coalesced over source rows.
__global__ void __launch_bounds__(256)
pack_weights_kernel(const float* __restrict__ W0, const float* __restrict__ W1,
                    const float* __restrict__ W2, const float* __restrict__ W3,
                    const float* __restrict__ Wfeat, const float* __restrict__ Wdir,
                    _Float16* __restrict__ dst) {
  const int m   = blockIdx.x / 6;
  const int seg = blockIdx.x % 6;
  const float* src; int N, K, KP, off;
  switch (seg) {
    case 0:  src = W0    + m * 63 * 128;  N = 128; K = 63;  KP = 64;  off = OFF_W0; break;
    case 1:  src = W1    + m * 128 * 128; N = 128; K = 128; KP = 128; off = OFF_W1; break;
    case 2:  src = W2    + m * 128 * 128; N = 128; K = 128; KP = 128; off = OFF_W2; break;
    case 3:  src = W3    + m * 128 * 128; N = 128; K = 128; KP = 128; off = OFF_W3; break;
    case 4:  src = Wfeat + m * 128 * 128; N = 128; K = 128; KP = 128; off = OFF_WF; break;
    default: src = Wdir  + m * 155 * 64;  N = 64;  K = 155; KP = 160; off = OFF_WD; break;
  }
  _Float16* d = dst + m * MLP_PACK + off;
  for (int j = threadIdx.x; j < K * N; j += 256) {         // coalesced read
    const int k = j / N, n = j % N;
    d[n * KP + k] = (_Float16)src[j];
  }
  for (int j = threadIdx.x; j < (KP - K) * N; j += 256) {  // zero K-padding
    const int n = j % N, k = K + j / N;
    d[n * KP + k] = (_Float16)0.0f;
  }
}

// ------------------------------ MLP kernel ----------------------------------
// One block = 128 samples, full 8-layer NeRF MLP with LDS-resident activations.
__global__ void __launch_bounds__(128)
nerf_mlp_kernel(const float* __restrict__ org, const float* __restrict__ dir,
                const float* __restrict__ tarr, int nsamp,
                const _Float16* __restrict__ wpk,
                const float* __restrict__ b0, const float* __restrict__ b1,
                const float* __restrict__ b2, const float* __restrict__ b3,
                const float* __restrict__ Wsig, const float* __restrict__ bsig,
                const float* __restrict__ bfeat, const float* __restrict__ bdir,
                const float* __restrict__ Wrgb, const float* __restrict__ brgb,
                float* __restrict__ preds) {
  extern __shared__ _Float16 smem_h[];
  const int tid = threadIdx.x;
  const int s0  = blockIdx.x * 128;

  // warm near caches with this MLP's whole f16 pack (global_prefetch_b8)
  for (int off = tid * 128; off < MLP_PACK * 2; off += 128 * 128)
    __builtin_prefetch((const char*)wpk + off, 0, 3);

  // ---- positional / direction encoding into LDS (one thread per sample) ----
  {
    const int gid = s0 + tid;
    const int ray = gid / nsamp;
    const float t = tarr[gid];
    float p[3], d[3];
#pragma unroll
    for (int j = 0; j < 3; ++j) {
      d[j] = dir[ray * 3 + j];
      p[j] = org[ray * 3 + j] + d[j] * t;
    }
    const int aOff = SMEM_A + tid * 128;
    const int dOff = SMEM_D + tid * 32;
#pragma unroll
    for (int j = 0; j < 3; ++j) {
      smem_h[aOff + j] = (_Float16)p[j];
      smem_h[dOff + j] = (_Float16)d[j];
    }
    float f = 1.0f;
    for (int i = 0; i < 10; ++i) {
#pragma unroll
      for (int j = 0; j < 3; ++j) {
        smem_h[aOff + 3 + 6 * i + j]     = (_Float16)__sinf(f * p[j]);
        smem_h[aOff + 3 + 6 * i + 3 + j] = (_Float16)__cosf(f * p[j]);
      }
      f *= 2.0f;
    }
    smem_h[aOff + 63] = (_Float16)0.0f;                  // K-pad 63 -> 64
    f = 1.0f;
    for (int i = 0; i < 4; ++i) {
#pragma unroll
      for (int j = 0; j < 3; ++j) {
        smem_h[dOff + 3 + 6 * i + j]     = (_Float16)__sinf(f * d[j]);
        smem_h[dOff + 3 + 6 * i + 3 + j] = (_Float16)__cosf(f * d[j]);
      }
      f *= 2.0f;
    }
    for (int j = 27; j < 32; ++j) smem_h[dOff + j] = (_Float16)0.0f;  // pad->160
  }

  // ---- trunk: 4 x relu(dense) with ping-pong activation buffers ----
  dense_layer<2, 0, 128, true >(SMEM_A, 128, 0, 0, wpk + OFF_W0, b0, SMEM_B, 128);
  dense_layer<4, 0, 128, true >(SMEM_B, 128, 0, 0, wpk + OFF_W1, b1, SMEM_A, 128);
  dense_layer<4, 0, 128, true >(SMEM_A, 128, 0, 0, wpk + OFF_W2, b2, SMEM_B, 128);
  dense_layer<4, 0, 128, true >(SMEM_B, 128, 0, 0, wpk + OFF_W3, b3, SMEM_A, 128);

  // ---- sigma head (128 -> 1), scalar dot, h currently in region A ----
  float sigma = bsig[0];
  for (int k = 0; k < 128; ++k)
    sigma += (float)smem_h[SMEM_A + tid * 128 + k] * Wsig[k];

  // ---- feature layer (no relu), then dir layer on concat(feat, dirs_enc) ----
  dense_layer<4, 0, 128, false>(SMEM_A, 128, 0, 0,        wpk + OFF_WF, bfeat, SMEM_B, 128);
  dense_layer<4, 1, 64,  true >(SMEM_B, 128, SMEM_D, 32,  wpk + OFF_WD, bdir,  SMEM_A, 128);

  // ---- rgb head (64 -> 3), write raw preds [rgb, sigma] as one b128 ----
  {
    const int gid = s0 + tid;
    float4 o;
    float r[3];
#pragma unroll
    for (int c = 0; c < 3; ++c) {
      float acc = brgb[c];
      for (int k = 0; k < 64; ++k)
        acc += (float)smem_h[SMEM_A + tid * 128 + k] * Wrgb[k * 3 + c];
      r[c] = acc;
    }
    o.x = r[0]; o.y = r[1]; o.z = r[2]; o.w = sigma;
    *(float4*)(preds + gid * 4) = o;
  }
}

// --------------------------- volume rendering -------------------------------
__global__ void render_kernel(const float* __restrict__ preds,
                              const float* __restrict__ tarr, int nsamp,
                              float* __restrict__ rgb_out,
                              float* __restrict__ depth_out,
                              float* __restrict__ w_out) {
  const int ray = blockIdx.x * blockDim.x + threadIdx.x;
  if (ray >= 1024) return;
  float T = 1.0f, r0 = 0, r1 = 0, r2 = 0, dep = 0;
  for (int i = 0; i < nsamp; ++i) {
    const float t     = tarr[ray * nsamp + i];
    const float delta = (i + 1 < nsamp) ? (tarr[ray * nsamp + i + 1] - t) : 1e10f;
    const float* p    = preds + (ray * nsamp + i) * 4;
    const float sg    = p[3] > 0.0f ? p[3] : 0.0f;
    const float alpha = 1.0f - __expf(-sg * delta);
    const float w     = alpha * T;
    r0  += w / (1.0f + __expf(-p[0]));
    r1  += w / (1.0f + __expf(-p[1]));
    r2  += w / (1.0f + __expf(-p[2]));
    dep += w * t;
    if (w_out) w_out[ray * nsamp + i] = w;
    T *= (1.0f - alpha + 1e-10f);
  }
  rgb_out[ray * 3 + 0] = r0;
  rgb_out[ray * 3 + 1] = r1;
  rgb_out[ray * 3 + 2] = r2;
  depth_out[ray] = dep;
}

// ---------------- hierarchical sampling: inverse-CDF + rank sort ------------
DEV float u01(unsigned x) {
  x ^= x >> 17; x *= 0xed5ad4bbu;
  x ^= x >> 11; x *= 0xac4c1b51u;
  x ^= x >> 15; x *= 0x31848babu;
  x ^= x >> 14;
  return (float)(x >> 8) * (1.0f / 16777216.0f);
}

__global__ void __launch_bounds__(192)
sample_sort_kernel(const float* __restrict__ t_vals,
                   const float* __restrict__ wts,
                   float* __restrict__ t_all) {
  __shared__ float cdf[64];     // 63 used
  __shared__ float bins[64];    // 63 midpoints
  __shared__ float vals[192];
  const int ray = blockIdx.x;
  const int tid = threadIdx.x;
  if (tid == 0) {
    float wl[62]; float s = 0.0f;
    for (int j = 0; j < 62; ++j) { wl[j] = wts[ray * 64 + 1 + j] + 1e-5f; s += wl[j]; }
    float c = 0.0f; cdf[0] = 0.0f;
    for (int j = 0; j < 62; ++j) { c += wl[j] / s; cdf[j + 1] = c; }
  }
  if (tid < 63) bins[tid] = 0.5f * (t_vals[ray * 64 + tid] + t_vals[ray * 64 + tid + 1]);
  if (tid < 64) vals[tid] = t_vals[ray * 64 + tid];
  __syncthreads();
  if (tid >= 64) {
    const int j = tid - 64;
    const float u = u01((unsigned)(ray * 131u + (unsigned)j * 7919u + 12345u));
    int ind = 0;
    for (int q = 0; q < 63; ++q) ind += (cdf[q] <= u) ? 1 : 0;   // searchsorted right
    int below = ind - 1; if (below < 0) below = 0;
    int above = ind;     if (above > 62) above = 62;
    const float cb = cdf[below], ca = cdf[above];
    float den = ca - cb; if (den < 1e-5f) den = 1.0f;
    const float tt = (u - cb) / den;
    vals[tid] = bins[below] + tt * (bins[above] - bins[below]);
  }
  __syncthreads();
  const float v = vals[tid];
  int rank = 0;
  for (int q = 0; q < 192; ++q) {
    const float vq = vals[q];
    rank += (vq < v || (vq == v && q < tid)) ? 1 : 0;
  }
  t_all[ray * 192 + rank] = v;
}

// ------------------------------- launcher -----------------------------------
extern "C" void kernel_launch(void* const* d_in, const int* in_sizes, int n_in,
                              void* d_out, int out_size, void* d_ws, size_t ws_size,
                              hipStream_t stream) {
  const float* org   = (const float*)d_in[0];
  const float* dirp  = (const float*)d_in[1];
  const float* tv    = (const float*)d_in[2];
  const float* W0    = (const float*)d_in[3];
  const float* b0    = (const float*)d_in[4];
  const float* W1    = (const float*)d_in[5];
  const float* b1    = (const float*)d_in[6];
  const float* W2    = (const float*)d_in[7];
  const float* b2    = (const float*)d_in[8];
  const float* W3    = (const float*)d_in[9];
  const float* b3    = (const float*)d_in[10];
  const float* Wsig  = (const float*)d_in[11];
  const float* bsig  = (const float*)d_in[12];
  const float* Wfeat = (const float*)d_in[13];
  const float* bfeat = (const float*)d_in[14];
  const float* Wdir  = (const float*)d_in[15];
  const float* bdir  = (const float*)d_in[16];
  const float* Wrgb  = (const float*)d_in[17];
  const float* brgb  = (const float*)d_in[18];

  float* out = (float*)d_out;
  float* ws  = (float*)d_ws;
  // workspace layout (float units)
  float*     preds_c = ws;                         // 1024*64*4   = 262144
  float*     wts_c   = ws + 262144;                // 1024*64     =  65536
  float*     t_all   = ws + 327680;                // 1024*192    = 196608
  float*     preds_f = ws + 524288;                // 1024*192*4  = 786432
  _Float16*  wpack   = (_Float16*)(ws + 1310720);  // 2*83968 halves

  // output layout: rgb_c[1024*3] | rgb_f[1024*3] | depth_c[1024] | depth_f[1024]
  float* rgb_c   = out;
  float* rgb_f   = out + 3072;
  float* depth_c = out + 6144;
  float* depth_f = out + 7168;

  constexpr size_t SMEM = SMEM_HALVES * sizeof(_Float16);  // 106496 B

  // ---- pack all weights to f16-transposed once (L2-resident afterwards) ----
  pack_weights_kernel<<<12, 256, 0, stream>>>(W0, W1, W2, W3, Wfeat, Wdir, wpack);

  // ---- coarse pass: MLP 0 over 1024*64 samples ----
  nerf_mlp_kernel<<<512, 128, SMEM, stream>>>(
      org, dirp, tv, 64, wpack,
      b0, b1, b2, b3, Wsig, bsig, bfeat, bdir, Wrgb, brgb, preds_c);
  render_kernel<<<8, 128, 0, stream>>>(preds_c, tv, 64, rgb_c, depth_c, wts_c);

  // ---- hierarchical sampling + merge-sort of 64+128 t values per ray ----
  sample_sort_kernel<<<1024, 192, 0, stream>>>(tv, wts_c, t_all);

  // ---- fine pass: MLP 1 over 1024*192 samples ----
  nerf_mlp_kernel<<<1536, 128, SMEM, stream>>>(
      org, dirp, t_all, 192, wpack + MLP_PACK,
      b0 + 128, b1 + 128, b2 + 128, b3 + 128,
      Wsig + 128, bsig + 1, bfeat + 128, bdir + 64,
      Wrgb + 64 * 3, brgb + 3, preds_f);
  render_kernel<<<8, 128, 0, stream>>>(preds_f, t_all, 192, rgb_f, depth_f, nullptr);
}
